// InteractionBlock_3985729650837
// MI455X (gfx1250) — compile-verified, matrix operand order checked by reading, not run on hardware
//
#include <hip/hip_runtime.h>
#include <math.h>

// ---------------------------------------------------------------------------
// GemNet InteractionBlock for MI455X (gfx1250, wave32, WMMA, wave-async copy)
// - All D x D GEMMs via v_wmma_f32_16x16x32_bf16 (K templated: 16/32/128)
// - All workspace intermediates stored bf16 (halves HBM traffic; workload is
//   slightly memory-bound: ~2.4 GB @ 23.3 TB/s ~ 105us vs ~60-90us compute)
// - Weights staged to LDS with GLOBAL_LOAD_ASYNC_TO_LDS_B128 when available,
//   then transposed+converted to bf16 [n][k] with bank-spread pitch
// ---------------------------------------------------------------------------

typedef __attribute__((ext_vector_type(16))) __bf16 v16bf;
typedef __attribute__((ext_vector_type(8)))  float  v8f;
typedef int v4i __attribute__((__vector_size__(16)));
typedef __attribute__((address_space(1))) v4i* gv4i_p;   // global int4*
typedef __attribute__((address_space(3))) v4i* lv4i_p;   // LDS int4*

#if defined(__has_builtin)
#if __has_builtin(__builtin_amdgcn_global_load_async_to_lds_b128)
#define HAS_ASYNC_LDS 1
#endif
#endif
#ifndef HAS_ASYNC_LDS
#define HAS_ASYNC_LDS 0
#endif

__device__ __forceinline__ unsigned short f2bf(float x) {
    union { float f; unsigned u; } a; a.f = x;
    unsigned r = a.u + 0x7FFFu + ((a.u >> 16) & 1u);   // round-to-nearest-even
    return (unsigned short)(r >> 16);
}
__device__ __forceinline__ float bf2f(unsigned short b) {
    union { unsigned u; float f; } v; v.u = (unsigned)b << 16; return v.f;
}
__device__ __forceinline__ unsigned pkbf(float a, float b) {
    return (unsigned)f2bf(a) | ((unsigned)f2bf(b) << 16);
}
__device__ __forceinline__ float silu_f(float x) {
    return x / (1.0f + __expf(-x));
}

struct AFrag { union { v16bf v; unsigned u[8]; uint4 q[2]; }; };
struct BFrag { union { v16bf v; uint4 q[2]; }; };

// out[M,128](bf16) = ((silu?)(X'@W + cin) * gout + addp) * (cscale * (*sptr))
// X'[r,k] = X[rowidx? rowidx[r] : r, k] * (gin? gin[r,k]:1) * in_scale
// X is bf16 (XBF) or fp32; W is K x 128 row-major fp32; M multiple of 128.
template<int K, bool XBF>
__global__ __launch_bounds__(256)
void k_gemm_t(const void* __restrict__ Xv, const int* __restrict__ rowidx,
              const unsigned short* __restrict__ gin, float in_scale,
              const float* __restrict__ Wg,
              const unsigned short* __restrict__ cin,
              const unsigned short* __restrict__ gout,
              const unsigned short* __restrict__ addp,
              const float* __restrict__ sptr, float cscale, int do_silu,
              unsigned short* __restrict__ out)
{
    constexpr int KP    = (K + 31) & ~31;       // K padded to WMMA K-step
    constexpr int KS    = KP >> 5;              // number of 32-wide K steps
    constexpr int PITCH = KP + 8;               // half-words per LDS weight row

    __shared__ __align__(16) unsigned short sW[128 * PITCH];
#if HAS_ASYNC_LDS
    __shared__ __align__(16) float sraw[K * 128];
#endif

    const int tid  = threadIdx.x;
    const int lane = tid & 31;
    const int wave = tid >> 5;

#if HAS_ASYNC_LDS
    // async DMA raw fp32 weights into LDS (16B per lane per op)
    for (int c = tid; c < (K * 128) / 4; c += 256) {
        __builtin_amdgcn_global_load_async_to_lds_b128(
            (gv4i_p)(Wg + c * 4), (lv4i_p)(sraw + c * 4), 0, 0);
    }
#if __has_builtin(__builtin_amdgcn_s_wait_asynccnt)
    __builtin_amdgcn_s_wait_asynccnt(0);
#else
    asm volatile("s_wait_asynccnt 0x0" ::: "memory");
#endif
    __syncthreads();
    const float* __restrict__ Wsrc = sraw;
#else
    const float* __restrict__ Wsrc = Wg;
#endif

    // zero-pad region [K,KP) (disjoint from transpose fill)
    if constexpr (KP > K) {
        constexpr int padw = KP - K;
        for (int i = tid; i < 128 * padw; i += 256) {
            int n = i / padw, k = K + (i % padw);
            sW[n * PITCH + k] = 0;
        }
    }
    // transpose + convert: sW[n][k] = bf16(W[k][n])
    for (int i = tid; i < K * 128; i += 256) {
        int k = i >> 7, n = i & 127;
        sW[n * PITCH + k] = f2bf(Wsrc[i]);
    }
    __syncthreads();

    const int r_off = lane & 15;
    const int half  = lane >> 4;
    const long row  = (long)blockIdx.x * 128 + wave * 16 + r_off;
    const long rs   = rowidx ? (long)rowidx[row] : row;

    // ---- build all A fragments up front (loads issue as one clause) ----
    AFrag af[KS];
    const bool amod = (gin != nullptr) || (in_scale != 1.0f);
#pragma unroll
    for (int s = 0; s < KS; ++s) {
#pragma unroll
        for (int c = 0; c < 2; ++c) {
            const int kk = s * 32 + half * 8 + c * 16;
            if (s * 32 + c * 16 < K) {          // compile-time (lane-indep.)
                if constexpr (XBF) {
                    const unsigned short* Xh = (const unsigned short*)Xv;
                    uint4 xq = *(const uint4*)(Xh + rs * (long)K + kk);
                    if (amod) {
                        union { uint4 q; unsigned w[4]; } xu, gu;
                        xu.q = xq;
                        gu.w[0] = gu.w[1] = gu.w[2] = gu.w[3] = 0;
                        if (gin)
                            gu.q = *(const uint4*)(gin + rs * (long)K + kk);
#pragma unroll
                        for (int j = 0; j < 4; ++j) {
                            float lo = bf2f((unsigned short)xu.w[j]) * in_scale;
                            float hi = bf2f((unsigned short)(xu.w[j] >> 16)) * in_scale;
                            if (gin) {
                                lo *= bf2f((unsigned short)gu.w[j]);
                                hi *= bf2f((unsigned short)(gu.w[j] >> 16));
                            }
                            af[s].u[c * 4 + j] = pkbf(lo, hi);
                        }
                    } else {
                        af[s].q[c] = xq;
                    }
                } else {
                    const float* Xf = (const float*)Xv;
                    const float4* p = (const float4*)(Xf + rs * (long)K + kk);
                    float4 a0 = p[0], a1 = p[1];
                    af[s].u[c * 4 + 0] = pkbf(a0.x * in_scale, a0.y * in_scale);
                    af[s].u[c * 4 + 1] = pkbf(a0.z * in_scale, a0.w * in_scale);
                    af[s].u[c * 4 + 2] = pkbf(a1.x * in_scale, a1.y * in_scale);
                    af[s].u[c * 4 + 3] = pkbf(a1.z * in_scale, a1.w * in_scale);
                }
            } else {
                af[s].u[c * 4 + 0] = 0; af[s].u[c * 4 + 1] = 0;
                af[s].u[c * 4 + 2] = 0; af[s].u[c * 4 + 3] = 0;
            }
        }
    }

    // ---- WMMA main: 8 N-tiles x KS K-steps ----
    v8f acc[8] = {};
#pragma unroll
    for (int s = 0; s < KS; ++s) {
        const int k0 = s * 32 + half * 8;
#pragma unroll
        for (int t = 0; t < 8; ++t) {
            BFrag bfr;
            const unsigned short* bp = sW + (t * 16 + r_off) * PITCH + k0;
            bfr.q[0] = *(const uint4*)bp;
            bfr.q[1] = *(const uint4*)(bp + 16);
            acc[t] = __builtin_amdgcn_wmma_f32_16x16x32_bf16(
                false, af[s].v, false, bfr.v, (short)0, acc[t], false, false);
        }
    }

    // ---- epilogue ----
    const float sval  = cscale * (sptr ? sptr[0] : 1.0f);
    const long  obase = ((long)blockIdx.x * 128 + wave * 16 + half * 8) * 128;
#pragma unroll
    for (int t = 0; t < 8; ++t) {
        const int col = t * 16 + r_off;
#pragma unroll
        for (int v = 0; v < 8; ++v) {
            const long idx = obase + (long)v * 128 + col;
            float a = acc[t][v];
            if (cin)     a += bf2f(cin[idx]);
            if (do_silu) a  = silu_f(a);
            if (gout)    a *= bf2f(gout[idx]);
            if (addp)    a += bf2f(addp[idx]);
            out[idx] = f2bf(a * sval);
        }
    }
}

// ---------------- elementwise helpers (dtype-flexible, 4-wide) -------------

__device__ __forceinline__ float ldx(const void* p, long i, int bf) {
    return bf ? bf2f(((const unsigned short*)p)[i]) : ((const float*)p)[i];
}
__device__ __forceinline__ void stx(void* p, long i, int bf, float v) {
    if (bf) ((unsigned short*)p)[i] = f2bf(v);
    else    ((float*)p)[i] = v;
}

__global__ void k_ew_add2(const void* __restrict__ a, int abf,
                          const void* __restrict__ b, int bbf, float s,
                          void* __restrict__ o, int obf, long n)
{
    for (long g = blockIdx.x * (long)blockDim.x + threadIdx.x; g * 4 < n;
         g += (long)gridDim.x * blockDim.x) {
        const long base = g * 4;
#pragma unroll
        for (int j = 0; j < 4; ++j)
            stx(o, base + j, obf, (ldx(a, base + j, abf) + ldx(b, base + j, bbf)) * s);
    }
}

__global__ void k_ew_add3(const unsigned short* __restrict__ a,
                          const unsigned short* __restrict__ b,
                          const unsigned short* __restrict__ c, float s,
                          unsigned short* __restrict__ o, long n)
{
    for (long g = blockIdx.x * (long)blockDim.x + threadIdx.x; g * 4 < n;
         g += (long)gridDim.x * blockDim.x) {
        const long base = g * 4;
#pragma unroll
        for (int j = 0; j < 4; ++j)
            o[base + j] = f2bf((bf2f(a[base + j]) + bf2f(b[base + j]) + bf2f(c[base + j])) * s);
    }
}

// o[e,:] = (a[e,:] + b[swap[e],:]) * s   (rows of 128)
__global__ void k_ew_swap(const unsigned short* __restrict__ a,
                          const unsigned short* __restrict__ b,
                          const int* __restrict__ sw, float s,
                          unsigned short* __restrict__ o, long n)
{
    for (long g = blockIdx.x * (long)blockDim.x + threadIdx.x; g * 4 < n;
         g += (long)gridDim.x * blockDim.x) {
        const long base = g * 4;
        const long e = base >> 7, cc = base & 127;
        const long bb = (long)sw[e] * 128 + cc;
#pragma unroll
        for (int j = 0; j < 4; ++j)
            o[base + j] = f2bf((bf2f(a[base + j]) + bf2f(b[bb + j])) * s);
    }
}

__global__ void k_zero(float* __restrict__ o, long n4)
{
    float4 z; z.x = z.y = z.z = z.w = 0.0f;
    for (long i = blockIdx.x * (long)blockDim.x + threadIdx.x; i < n4;
         i += (long)gridDim.x * blockDim.x)
        ((float4*)o)[i] = z;
}

// segment-sum: o[idx[e],c] += m[e,c]*g[e,c]*(*sptr)  (fp32 global atomics)
__global__ void k_scatter(const unsigned short* __restrict__ m,
                          const unsigned short* __restrict__ g,
                          const float* __restrict__ sptr,
                          const int* __restrict__ idx,
                          float* __restrict__ o, long n)
{
    const float s = sptr[0];
    for (long q = blockIdx.x * (long)blockDim.x + threadIdx.x; q * 4 < n;
         q += (long)gridDim.x * blockDim.x) {
        const long base = q * 4;
        const long e = base >> 7, cc = base & 127;
        float* dst = o + (long)idx[e] * 128 + cc;
#pragma unroll
        for (int j = 0; j < 4; ++j)
            __hip_atomic_fetch_add(dst + j, bf2f(m[base + j]) * bf2f(g[base + j]) * s,
                                   __ATOMIC_RELAXED, __HIP_MEMORY_SCOPE_AGENT);
    }
}

// ---------------------------------------------------------------------------

typedef unsigned short bfh;

static inline void G(hipStream_t st, long M, const void* X, int xbf, const int* ri,
                     const bfh* gin, float insc, int K, const float* W,
                     const bfh* cin, const bfh* gout, const bfh* addp,
                     const float* sptr, float cs, int silu, bfh* out)
{
    dim3 gr((unsigned)(M / 128)), bl(256);
    if (K == 128) {
        if (xbf) k_gemm_t<128, true ><<<gr, bl, 0, st>>>(X, ri, gin, insc, W, cin, gout, addp, sptr, cs, silu, out);
        else     k_gemm_t<128, false><<<gr, bl, 0, st>>>(X, ri, gin, insc, W, cin, gout, addp, sptr, cs, silu, out);
    } else if (K == 32) {
        k_gemm_t<32, false><<<gr, bl, 0, st>>>(X, ri, gin, insc, W, cin, gout, addp, sptr, cs, silu, out);
    } else {
        k_gemm_t<16, false><<<gr, bl, 0, st>>>(X, ri, gin, insc, W, cin, gout, addp, sptr, cs, silu, out);
    }
}
static inline unsigned ewg(long groups) {
    long b = (groups + 255) / 256;
    return (unsigned)(b > 65535 ? 65535 : b);
}

extern "C" void kernel_launch(void* const* d_in, const int* in_sizes, int n_in,
                              void* d_out, int out_size, void* d_ws, size_t ws_size,
                              hipStream_t stream)
{
    (void)n_in; (void)out_size; (void)ws_size;
    const float INV2 = 0.70710678118654752f;
    const float INV3 = 0.57735026918962576f;

    const float* h     = (const float*)d_in[0];
    const float* m_ij  = (const float*)d_in[1];
    const float* rbf_h = (const float*)d_in[2];
    const float* rbf   = (const float*)d_in[3];
    const float* rbfp  = (const float*)d_in[4];
    const float* cbf   = (const float*)d_in[5];
    const float* cbfp  = (const float*)d_in[6];
    const float* sbf   = (const float*)d_in[7];
    const int*   idx_i = (const int*)d_in[8];
    const int*   idx_j = (const int*)d_in[9];
    const int*   idxsw = (const int*)d_in[10];

    const float* W_mlp_ij = (const float*)d_in[11];
    const float* qWm_rbf = (const float*)d_in[12];
    const float* qW_rbf  = (const float*)d_in[13];
    const float* qWm_cbf = (const float*)d_in[14];
    const float* qW_cbf  = (const float*)d_in[15];
    const float* qWm_sbf = (const float*)d_in[16];
    const float* qW_sbf  = (const float*)d_in[17];
    const float* qW_dir  = (const float*)d_in[18];
    const float* qW_ij   = (const float*)d_in[19];
    const float* qW_ji   = (const float*)d_in[20];
    const float* q_srbf  = (const float*)d_in[21];
    const float* q_scbf  = (const float*)d_in[22];
    const float* q_ssbf  = (const float*)d_in[23];
    const float* tWm_rbf = (const float*)d_in[24];
    const float* tW_rbf  = (const float*)d_in[25];
    const float* tWm_cbf = (const float*)d_in[26];
    const float* tW_cbf  = (const float*)d_in[27];
    const float* tW_dir  = (const float*)d_in[28];
    const float* tW_ij   = (const float*)d_in[29];
    const float* tW_ji   = (const float*)d_in[30];
    const float* t_srbf  = (const float*)d_in[31];
    const float* t_scbf  = (const float*)d_in[32];
    const float* res_bef = (const float*)d_in[33];   // (1,2,128,128)
    const float* res_aft = (const float*)d_in[34];   // (2,2,128,128)
    const float* aeW_rbf = (const float*)d_in[35];
    const float* aeW_in  = (const float*)d_in[36];
    const float* ae_res  = (const float*)d_in[37];   // (2,2,128,128)
    const float* ae_ssum = (const float*)d_in[38];
    const float* W_self  = (const float*)d_in[39];   // (384,128)
    const float* res_m   = (const float*)d_in[40];   // (1,2,128,128)

    const long N = in_sizes[0] / 128;
    const long E = in_sizes[1] / 128;
    const long ED = E * 128, ND = N * 128;
    const long DD = 128 * 128;

    // workspace: 7 edge-sized bf16 streams, fp32 atom accumulator, 3 atom bf16
    bfh* B0 = (bfh*)d_ws;
    bfh* B1 = B0 + ED;  bfh* B2 = B1 + ED;  bfh* B3 = B2 + ED;
    bfh* B4 = B3 + ED;  bfh* B5 = B4 + ED;  bfh* B6 = B5 + ED;
    float* A0 = (float*)(B6 + ED);          // fp32 scatter accumulator
    bfh* A1 = (bfh*)(A0 + ND);
    bfh* A2 = A1 + ND;  bfh* A3 = A2 + ND;

    float* Hnew = (float*)d_out;            // N*128 fp32
    float* Mnew = Hnew + ND;                // E*128 fp32

    // ---------------- quad branch -> x4 in B0 ----------------
    G(stream, E, m_ij, 0, 0, 0, 1.f, 128, qWm_rbf, 0, 0, 0, 0, 1.f, 1, B2);   // m1q
    for (int nb = 0; nb < 2; ++nb) {
        const float* r = rbf + (long)nb * E * 16;
        const float* c = cbf + (long)nb * E * 16;
        const float* s = sbf + (long)nb * E * 32;
        G(stream, E, r, 0, 0, 0, 1.f, 16, qW_rbf, 0, 0, 0, q_srbf, 1.f, 0, B3); // g_rbf
        G(stream, E, B2, 1, 0, B3, 1.f, 128, qWm_cbf, 0, 0, 0, 0, 1.f, 1, B4);
        G(stream, E, c, 0, 0, 0, 1.f, 16, qW_cbf, 0, 0, 0, q_scbf, 1.f, 0, B3); // g_cbf
        G(stream, E, s, 0, 0, 0, 1.f, 32, qW_sbf, 0, 0, 0, q_ssbf, 1.f, 0, B6); // g_sbf
        G(stream, E, B4, 1, 0, B3, 1.f, 128, qWm_sbf, 0, B6, nb ? B0 : 0, 0, 1.f, 1, B0);
    }
    G(stream, E, B0, 1, 0, 0, INV2, 128, qW_dir, 0, 0, 0, 0, 1.f, 1, B4);     // x_dir
    G(stream, E, B4, 1, 0, 0, 1.f, 128, qW_ij, 0, 0, 0, 0, 1.f, 1, B5);       // x_ij
    G(stream, E, B4, 1, 0, 0, 1.f, 128, qW_ji, 0, 0, 0, 0, 1.f, 1, B6);       // x_ji
    k_ew_swap<<<ewg(ED / 4), 256, 0, stream>>>(B5, B6, idxsw, INV2, B0, ED);

    // ---------------- trip branch -> x3 in B1 ----------------
    G(stream, E, m_ij, 0, 0, 0, 1.f, 128, tWm_rbf, 0, 0, 0, 0, 1.f, 1, B2);   // m1t
    for (int nb = 0; nb < 2; ++nb) {
        const float* r = rbfp + (long)nb * E * 16;
        const float* c = cbfp + (long)nb * E * 16;
        G(stream, E, r, 0, 0, 0, 1.f, 16, tW_rbf, 0, 0, 0, t_srbf, 1.f, 0, B3);
        G(stream, E, c, 0, 0, 0, 1.f, 16, tW_cbf, 0, 0, 0, t_scbf, 1.f, 0, B6);
        G(stream, E, B2, 1, 0, B3, 1.f, 128, tWm_cbf, 0, B6, nb ? B1 : 0, 0, 1.f, 1, B1);
    }
    G(stream, E, B1, 1, 0, 0, INV2, 128, tW_dir, 0, 0, 0, 0, 1.f, 1, B4);
    G(stream, E, B4, 1, 0, 0, 1.f, 128, tW_ij, 0, 0, 0, 0, 1.f, 1, B5);
    G(stream, E, B4, 1, 0, 0, 1.f, 128, tW_ji, 0, 0, 0, 0, 1.f, 1, B6);
    k_ew_swap<<<ewg(ED / 4), 256, 0, stream>>>(B5, B6, idxsw, INV2, B1, ED);

    // ---------------- combine + residuals on edges ----------------
    G(stream, E, m_ij, 0, 0, 0, 1.f, 128, W_mlp_ij, 0, 0, 0, 0, 1.f, 0, B2);  // x_skip
    k_ew_add3<<<ewg(ED / 4), 256, 0, stream>>>(B0, B1, B2, INV3, B4, ED);     // x
    G(stream, E, B4, 1, 0, 0, 1.f, 128, res_bef, 0, 0, 0, 0, 1.f, 1, B5);
    G(stream, E, B5, 1, 0, 0, 1.f, 128, res_bef + DD, 0, 0, B4, 0, INV2, 1, B0);
    k_ew_add2<<<ewg(ED / 4), 256, 0, stream>>>(m_ij, 0, B0, 1, INV2, B1, 1, ED); // m
    G(stream, E, B1, 1, 0, 0, 1.f, 128, res_aft, 0, 0, 0, 0, 1.f, 1, B4);
    G(stream, E, B4, 1, 0, 0, 1.f, 128, res_aft + DD, 0, 0, B1, 0, INV2, 1, B2);
    G(stream, E, B2, 1, 0, 0, 1.f, 128, res_aft + 2 * DD, 0, 0, 0, 0, 1.f, 1, B4);
    G(stream, E, B4, 1, 0, 0, 1.f, 128, res_aft + 3 * DD, 0, 0, B2, 0, INV2, 1, B1); // m final

    // ---------------- atom embedding (scatter-add) ----------------
    G(stream, E, rbf_h, 0, 0, 0, 1.f, 16, aeW_rbf, 0, 0, 0, 0, 1.f, 0, B3);   // g_rbf_h
    k_zero<<<ewg(ND / 4), 256, 0, stream>>>(A0, ND / 4);
    k_scatter<<<ewg(ED / 4), 256, 0, stream>>>(B1, B3, ae_ssum, idx_i, A0, ED);
    G(stream, N, A0, 0, 0, 0, 1.f, 128, aeW_in, 0, 0, 0, 0, 1.f, 1, A1);
    G(stream, N, A1, 1, 0, 0, 1.f, 128, ae_res, 0, 0, 0, 0, 1.f, 1, A2);
    G(stream, N, A2, 1, 0, 0, 1.f, 128, ae_res + DD, 0, 0, A1, 0, INV2, 1, A3);
    G(stream, N, A3, 1, 0, 0, 1.f, 128, ae_res + 2 * DD, 0, 0, 0, 0, 1.f, 1, A2);
    G(stream, N, A2, 1, 0, 0, 1.f, 128, ae_res + 3 * DD, 0, 0, A3, 0, INV2, 1, A1);
    k_ew_add2<<<ewg(ND / 4), 256, 0, stream>>>(h, 0, A1, 1, INV2, Hnew, 0, ND); // h_new

    // ---------------- atom self-interaction (gathered concat GEMM) -------
    G(stream, E, Hnew, 0, idx_i, 0, 1.f, 128, W_self, 0, 0, 0, 0, 1.f, 0, B4);
    G(stream, E, Hnew, 0, idx_j, 0, 1.f, 128, W_self + DD, B4, 0, 0, 0, 1.f, 0, B4);
    G(stream, E, B1, 1, 0, 0, 1.f, 128, W_self + 2 * DD, B4, 0, 0, 0, 1.f, 1, B5); // m2
    G(stream, E, B5, 1, 0, 0, 1.f, 128, res_m, 0, 0, 0, 0, 1.f, 1, B4);
    G(stream, E, B4, 1, 0, 0, 1.f, 128, res_m + DD, 0, 0, B5, 0, INV2, 1, B2);
    k_ew_add2<<<ewg(ED / 4), 256, 0, stream>>>(B1, 1, B2, 1, INV2, Mnew, 0, ED); // m_new
}
